// SAE_23046794510385
// MI455X (gfx1250) — compile-verified, matrix operand order checked by reading, not run on hardware
//
#include <hip/hip_runtime.h>
#include <stdint.h>

typedef __attribute__((ext_vector_type(16))) __bf16 v16bf;
typedef __attribute__((ext_vector_type(8)))  __bf16 v8bf;
typedef __attribute__((ext_vector_type(8)))  float  v8f;

constexpr int N_ROWS = 8192;   // batch
constexpr int DM     = 1024;   // d_model (GEMM K)
constexpr int DS     = 32768;  // d_sae   (GEMM N)
constexpr int KTOP   = 100;

constexpr int BM = 128, BN = 128, BK = 32;
constexpr int BKP = BK + 8;    // LDS row pad (80B stride) to dodge bank conflicts

__device__ __forceinline__ __bf16 f2bf(float f) {
  unsigned u = __float_as_uint(f);
  u += 0x7FFFu + ((u >> 16) & 1u);           // round-to-nearest-even
  unsigned short h = (unsigned short)(u >> 16);
  __bf16 b;
  __builtin_memcpy(&b, &h, sizeof(h));
  return b;
}

__device__ __forceinline__ v16bf join16(v8bf lo, v8bf hi) {
  struct P { v8bf a, b; } p{lo, hi};
  v16bf r;
  __builtin_memcpy(&r, &p, sizeof(r));
  return r;
}

// ---------------------------------------------------------------------------
// Kernel 1: acts = relu((x - d_bias) @ W_enc^T + b_enc), bf16 WMMA
// grid = (DS/BN, N_ROWS/BM), block = 256 (8 waves; 2x4 wave grid, 64x32/wave)
// ---------------------------------------------------------------------------
__global__ __launch_bounds__(256)
void sae_encode_wmma(const float* __restrict__ x,
                     const float* __restrict__ W_enc,
                     const float* __restrict__ b_enc,
                     const float* __restrict__ d_bias,
                     float* __restrict__ acts /* [N_ROWS][DS] */)
{
  __shared__ __bf16 As[BM][BKP];
  __shared__ __bf16 Bs[BN][BKP];

  const int m0 = blockIdx.y * BM;
  const int n0 = blockIdx.x * BN;

  const int tid  = threadIdx.x;
  const int lane = tid & 31;
  const int wave = tid >> 5;
  const int wm   = wave >> 2;         // 0..1 : 64-row slab
  const int wn   = wave & 3;          // 0..3 : 32-col slab
  const int lr   = lane & 15;         // row/col within 16-tile
  const int hk   = (lane >> 4) * 8;   // K sub-chunk base: 0 or 8

  v8f acc[4][2] = {};

  for (int k0 = 0; k0 < DM; k0 += BK) {
    // Cooperative tile fill: fp32 global -> bf16 LDS (A gets -decoder_bias)
#pragma unroll
    for (int q = 0; q < 4; ++q) {
      int flat = tid + q * 256;                 // float4 index in 128x32 tile
      int r    = flat >> 3;                     // 8 float4 per 32-wide row
      int c    = (flat & 7) << 2;
      const float4 av = *(const float4*)(x     + (size_t)(m0 + r) * DM + k0 + c);
      const float4 bv = *(const float4*)(W_enc + (size_t)(n0 + r) * DM + k0 + c);
      const float4 db = *(const float4*)(d_bias + k0 + c);
      As[r][c + 0] = f2bf(av.x - db.x);
      As[r][c + 1] = f2bf(av.y - db.y);
      As[r][c + 2] = f2bf(av.z - db.z);
      As[r][c + 3] = f2bf(av.w - db.w);
      Bs[r][c + 0] = f2bf(bv.x);
      Bs[r][c + 1] = f2bf(bv.y);
      Bs[r][c + 2] = f2bf(bv.z);
      Bs[r][c + 3] = f2bf(bv.w);
    }
    __syncthreads();

    // Fragment loads: lane L<16 holds K {0..7,16..23}, L>=16 holds {8..15,24..31}
    v16bf afrag[4], bfrag[2];
#pragma unroll
    for (int t = 0; t < 4; ++t) {
      int r = wm * 64 + t * 16 + lr;
      afrag[t] = join16(*(const v8bf*)&As[r][hk], *(const v8bf*)&As[r][hk + 16]);
    }
#pragma unroll
    for (int t = 0; t < 2; ++t) {
      int r = wn * 32 + t * 16 + lr;
      bfrag[t] = join16(*(const v8bf*)&Bs[r][hk], *(const v8bf*)&Bs[r][hk + 16]);
    }

#pragma unroll
    for (int tm = 0; tm < 4; ++tm)
#pragma unroll
      for (int tn = 0; tn < 2; ++tn)
        acc[tm][tn] = __builtin_amdgcn_wmma_f32_16x16x32_bf16(
            false, afrag[tm], false, bfrag[tn],
            (short)0, acc[tm][tn], false, false);
    __syncthreads();
  }

  // Epilogue: +b_enc, ReLU, store. VGPR v holds M=v (lanes<16) / v+8 (lanes>=16)
#pragma unroll
  for (int tn = 0; tn < 2; ++tn) {
    int gcol = n0 + wn * 32 + tn * 16 + lr;
    float be = b_enc[gcol];
#pragma unroll
    for (int tm = 0; tm < 4; ++tm) {
      int rbase = m0 + wm * 64 + tm * 16 + (lane >> 4) * 8;
#pragma unroll
      for (int v = 0; v < 8; ++v) {
        float val = acc[tm][tn][v] + be;
        val = val > 0.0f ? val : 0.0f;
        acts[(size_t)(rbase + v) * DS + gcol] = val;
      }
    }
  }
}

// ---------------------------------------------------------------------------
// Kernel 2: per-row top-K via MSB-first radix select on nonneg float bits.
// Zeroes non-top-K in place; optionally emits compacted (idx,val) lists.
// grid = N_ROWS, block = 256
// ---------------------------------------------------------------------------
__global__ __launch_bounds__(256)
void sae_topk_radix(float* __restrict__ acts,
                    int* __restrict__ idx_out, float* __restrict__ val_out)
{
  const int n   = blockIdx.x;
  const int tid = threadIdx.x;
  float* row = acts + (size_t)n * DS;

  __shared__ int      hist[256];
  __shared__ unsigned sh_pre;
  __shared__ int      sh_rem, sh_tie, sh_out;

  if (tid == 0) { sh_pre = 0u; sh_rem = KTOP; sh_tie = 0; sh_out = 0; }
  __syncthreads();

  for (int d = 3; d >= 0; --d) {
    hist[tid] = 0;
    __syncthreads();
    const unsigned pre   = sh_pre;
    const unsigned shift = (unsigned)(d + 1) * 8u;
    for (int i = tid; i < DS; i += 256) {
      unsigned u = __float_as_uint(row[i]);
      if (((unsigned long long)(u ^ pre) >> shift) == 0ull)   // high digits match
        atomicAdd(&hist[(u >> (8 * d)) & 255], 1);
    }
    __syncthreads();
    if (tid == 0) {
      int rem = sh_rem;
      int b = 255;
      for (; b > 0; --b) {
        int c = hist[b];
        if (c >= rem) break;
        rem -= c;
      }
      sh_pre = pre | ((unsigned)b << (8 * d));
      sh_rem = rem;
    }
    __syncthreads();
  }

  const unsigned T   = sh_pre;   // bit pattern of K-th largest
  const int tiesNeed = sh_rem;   // how many ==T survive

  for (int i = tid; i < DS; i += 256) {
    float v = row[i];
    unsigned u = __float_as_uint(v);
    bool keep = false;
    if (u > T) keep = true;
    else if (u == T) keep = (atomicAdd(&sh_tie, 1) < tiesNeed);
    if (keep) {
      if (idx_out) {
        int s = atomicAdd(&sh_out, 1);
        if (s < KTOP) {
          idx_out[(size_t)n * KTOP + s] = i;
          val_out[(size_t)n * KTOP + s] = v;
        }
      }
    } else {
      row[i] = 0.0f;
    }
  }
}

// ---------------------------------------------------------------------------
// Kernel 3: W_dec [DM][DS] -> WdT [DS][DM] (L2-resident 128MB gather table)
// grid = (DS/32, DM/32), block = (32,8)
// ---------------------------------------------------------------------------
__global__ __launch_bounds__(256)
void wdec_transpose(const float* __restrict__ Wd, float* __restrict__ WdT)
{
  __shared__ float t[32][33];
  const int j0 = blockIdx.x * 32;
  const int m0 = blockIdx.y * 32;
  const int tx = threadIdx.x, ty = threadIdx.y;
#pragma unroll
  for (int r = ty; r < 32; r += 8)
    t[r][tx] = Wd[(size_t)(m0 + r) * DS + j0 + tx];
  __syncthreads();
#pragma unroll
  for (int r = ty; r < 32; r += 8)
    WdT[(size_t)(j0 + r) * DM + m0 + tx] = t[tx][r];
}

// ---------------------------------------------------------------------------
// Kernel 4: sparse decode y[n] = d_bias + sum_k val*WdT[j]  (coalesced rows)
// grid = N_ROWS, block = 256 (each thread owns 4 of the 1024 output dims)
// ---------------------------------------------------------------------------
__global__ __launch_bounds__(256)
void sae_decode_gather(const float* __restrict__ WdT,
                       const float* __restrict__ d_bias,
                       const int* __restrict__ idx_l,
                       const float* __restrict__ val_l,
                       float* __restrict__ y)
{
  const int n = blockIdx.x, tid = threadIdx.x;
  float acc[4];
#pragma unroll
  for (int q = 0; q < 4; ++q) acc[q] = d_bias[tid + q * 256];
  for (int k = 0; k < KTOP; ++k) {
    int   j = idx_l[(size_t)n * KTOP + k];
    float v = val_l[(size_t)n * KTOP + k];
    const float* wr = WdT + (size_t)j * DM;
#pragma unroll
    for (int q = 0; q < 4; ++q) acc[q] += v * wr[tid + q * 256];
  }
#pragma unroll
  for (int q = 0; q < 4; ++q) y[(size_t)n * DM + tid + q * 256] = acc[q];
}

// Fallback if workspace can't hold WdT: strided gather straight from W_dec
__global__ __launch_bounds__(256)
void sae_decode_strided(const float* __restrict__ Wd,
                        const float* __restrict__ d_bias,
                        const int* __restrict__ idx_l,
                        const float* __restrict__ val_l,
                        float* __restrict__ y)
{
  const int n = blockIdx.x, tid = threadIdx.x;
  float acc[4];
#pragma unroll
  for (int q = 0; q < 4; ++q) acc[q] = d_bias[tid + q * 256];
  for (int k = 0; k < KTOP; ++k) {
    int   j = idx_l[(size_t)n * KTOP + k];
    float v = val_l[(size_t)n * KTOP + k];
#pragma unroll
    for (int q = 0; q < 4; ++q) acc[q] += v * Wd[(size_t)(tid + q * 256) * DS + j];
  }
#pragma unroll
  for (int q = 0; q < 4; ++q) y[(size_t)n * DM + tid + q * 256] = acc[q];
}

// Fallback if workspace can't even hold the lists: scan latents row directly
__global__ __launch_bounds__(256)
void sae_decode_scan(const float* __restrict__ Wd,
                     const float* __restrict__ d_bias,
                     const float* __restrict__ latents,
                     float* __restrict__ y)
{
  const int n = blockIdx.x, tid = threadIdx.x;
  const float* row = latents + (size_t)n * DS;
  float acc[4];
#pragma unroll
  for (int q = 0; q < 4; ++q) acc[q] = d_bias[tid + q * 256];
  for (int j = 0; j < DS; ++j) {
    float v = row[j];
    if (v != 0.0f) {
#pragma unroll
      for (int q = 0; q < 4; ++q) acc[q] += v * Wd[(size_t)(tid + q * 256) * DS + j];
    }
  }
#pragma unroll
  for (int q = 0; q < 4; ++q) y[(size_t)n * DM + tid + q * 256] = acc[q];
}

extern "C" void kernel_launch(void* const* d_in, const int* in_sizes, int n_in,
                              void* d_out, int out_size, void* d_ws, size_t ws_size,
                              hipStream_t stream)
{
  (void)in_sizes; (void)n_in; (void)out_size;

  const float* x      = (const float*)d_in[0];
  const float* W_enc  = (const float*)d_in[1];
  const float* b_enc  = (const float*)d_in[2];
  const float* W_dec  = (const float*)d_in[3];
  const float* d_bias = (const float*)d_in[4];

  float* y       = (float*)d_out;                    // [N_ROWS][DM]
  float* latents = y + (size_t)N_ROWS * DM;          // [N_ROWS][DS]

  const size_t listBytes = (size_t)N_ROWS * KTOP * (sizeof(int) + sizeof(float));
  const size_t off       = (listBytes + 255) & ~(size_t)255;
  const bool haveLists = ws_size >= listBytes;
  const bool haveT     = ws_size >= off + (size_t)DM * DS * sizeof(float);

  int*   idx_l = haveLists ? (int*)d_ws : nullptr;
  float* val_l = haveLists ? (float*)((char*)d_ws + (size_t)N_ROWS * KTOP * sizeof(int))
                           : nullptr;
  float* WdT   = (float*)((char*)d_ws + off);

  // 1) Encoder GEMM + bias + ReLU -> latents region of d_out (acts)
  dim3 ggrid(DS / BN, N_ROWS / BM);
  sae_encode_wmma<<<ggrid, 256, 0, stream>>>(x, W_enc, b_enc, d_bias, latents);

  // 2) In-place top-K sparsification + (idx,val) list emission
  sae_topk_radix<<<N_ROWS, 256, 0, stream>>>(latents, idx_l, val_l);

  // 3/4) Sparse decode
  if (haveT && haveLists) {
    dim3 tgrid(DS / 32, DM / 32), tblk(32, 8);
    wdec_transpose<<<tgrid, tblk, 0, stream>>>(W_dec, WdT);
    sae_decode_gather<<<N_ROWS, 256, 0, stream>>>(WdT, d_bias, idx_l, val_l, y);
  } else if (haveLists) {
    sae_decode_strided<<<N_ROWS, 256, 0, stream>>>(W_dec, d_bias, idx_l, val_l, y);
  } else {
    sae_decode_scan<<<N_ROWS, 256, 0, stream>>>(W_dec, d_bias, latents, y);
  }
}
